// ExactAttention_90675349553510
// MI455X (gfx1250) — compile-verified, matrix operand order checked by reading, not run on hardware
//
#include <hip/hip_runtime.h>

typedef __attribute__((ext_vector_type(16))) _Float16 v16h;
typedef __attribute__((ext_vector_type(8)))  _Float16 v8h;
typedef __attribute__((ext_vector_type(8)))  float    v8f;

#define B_  2
#define N_  2048
#define H_  16
#define D_  128
#define BN  128              // Q rows per workgroup, keys per K/V tile
#define NT  (N_ / BN)        // 16 key tiles (even -> 2-tile unrolled loop)
#define KS  136              // LDS row stride in halfs (272B: 16B-aligned, bank-friendly)
#define TILE_H (BN * KS)     // halfs per LDS tile buffer
#define STEP ((size_t)BN * H_ * D_)   // floats between consecutive key tiles
#define SMEM_BYTES (5 * TILE_H * 2)   // K0,K1,V0,V1,P  ~170KB

// Load a 16-wide x 32-deep f16 fragment (A or B layout, mirrored):
// lane&15 selects row/col n; lane>>4 selects the K-half.
// V0..V3 <- K = 8*half + 0..7 ; V4..V7 <- K = 16 + 8*half + 0..7
// Source must be row-major with the fragment's K dimension contiguous.
__device__ __forceinline__ v16h load_frag16(const _Float16* tile, int lane, int kbase) {
    const int n  = lane & 15;
    const int hl = (lane >> 4) & 1;
    const _Float16* p = tile + n * KS + kbase + 8 * hl;
    v8h lo = *(const v8h*)(p);
    v8h hi = *(const v8h*)(p + 16);
    return __builtin_shufflevector(lo, hi, 0,1,2,3,4,5,6,7,8,9,10,11,12,13,14,15);
}

// Issue global loads for one K/V tile into register temps.
__device__ __forceinline__ void load_tile(const float* kp, const float* vp0,
                                          float4* kreg, float4* va, float4* vb) {
#pragma unroll
    for (int i = 0; i < 16; ++i) kreg[i] = *(const float4*)(kp + 4 * i);
    const float* vp1 = vp0 + H_ * D_;
#pragma unroll
    for (int i = 0; i < 8; ++i) {
        va[i] = *(const float4*)(vp0 + 4 * i);
        vb[i] = *(const float4*)(vp1 + 4 * i);
    }
}

// Convert register temps to f16 and store into an LDS tile buffer.
// K goes row-major [key][KS]; V goes transposed [dim][KS].
__device__ __forceinline__ void store_tile(_Float16* dk, _Float16* dvbase,
                                           int g, int kpair,
                                           const float4* kreg, const float4* va, const float4* vb) {
#pragma unroll
    for (int i = 0; i < 16; ++i) {
        union { _Float16 h[4]; uint2 u; } t2;
        t2.h[0] = (_Float16)kreg[i].x; t2.h[1] = (_Float16)kreg[i].y;
        t2.h[2] = (_Float16)kreg[i].z; t2.h[3] = (_Float16)kreg[i].w;
        *(uint2*)(dk + 4 * i) = t2.u;
    }
#pragma unroll
    for (int i = 0; i < 8; ++i) {
        const float af[4] = {va[i].x, va[i].y, va[i].z, va[i].w};
        const float bf[4] = {vb[i].x, vb[i].y, vb[i].z, vb[i].w};
#pragma unroll
        for (int e = 0; e < 4; ++e) {
            union { _Float16 h[2]; unsigned u; } t2;
            t2.h[0] = (_Float16)af[e];
            t2.h[1] = (_Float16)bf[e];
            *(unsigned*)(dvbase + (32 * g + 4 * i + e) * KS + 2 * kpair) = t2.u;
        }
    }
}

// One flash-attention step over a 128-key tile already staged in LDS (cK, cV).
__device__ __forceinline__ void process_tile(const _Float16* cK, const _Float16* cV,
                                             _Float16* pw, int lane, int nl, int hl,
                                             const v16h* Qa, v8f* O,
                                             float* mrow, float* lrow) {
    const v8f vzero = {0.f,0.f,0.f,0.f,0.f,0.f,0.f,0.f};

    // ---- S = Q (16xD) x K^T (Dx128) ----
    v8f S[8];
#pragma unroll
    for (int t = 0; t < 8; ++t) {
        v8f s = vzero;
#pragma unroll
        for (int c = 0; c < 4; ++c) {
            v16h bfrag = load_frag16(cK + 16 * t * KS, lane, 32 * c);
            s = __builtin_amdgcn_wmma_f32_16x16x32_f16(
                    false, Qa[c], false, bfrag, (short)0, s, false, false);
        }
        S[t] = s;
    }

    // ---- Online softmax (rows r+8*hl handled per lane-half) ----
#pragma unroll
    for (int r = 0; r < 8; ++r) {
        float x = S[0][r];
#pragma unroll
        for (int t = 1; t < 8; ++t) x = fmaxf(x, S[t][r]);
        x = fmaxf(x, __shfl_xor(x, 1, 16));
        x = fmaxf(x, __shfl_xor(x, 2, 16));
        x = fmaxf(x, __shfl_xor(x, 4, 16));
        x = fmaxf(x, __shfl_xor(x, 8, 16));
        const float nm    = fmaxf(mrow[r], x);
        const float alpha = __expf(mrow[r] - nm);
        float rs = 0.f;
#pragma unroll
        for (int t = 0; t < 8; ++t) {
            float p = __expf(S[t][r] - nm);
            S[t][r] = p;
            rs += p;
        }
        rs += __shfl_xor(rs, 1, 16);
        rs += __shfl_xor(rs, 2, 16);
        rs += __shfl_xor(rs, 4, 16);
        rs += __shfl_xor(rs, 8, 16);
        lrow[r] = lrow[r] * alpha + rs;
        mrow[r] = nm;
#pragma unroll
        for (int t = 0; t < 8; ++t) O[t][r] *= alpha;
    }

    // ---- P (C-layout) -> per-wave LDS, f16 ----
#pragma unroll
    for (int t = 0; t < 8; ++t)
#pragma unroll
        for (int r = 0; r < 8; ++r)
            pw[(r + 8 * hl) * KS + 16 * t + nl] = (_Float16)S[t][r];

    // ---- O += P (16x128) x V (128xD) ----
#pragma unroll
    for (int c = 0; c < 4; ++c) {
        v16h pa = load_frag16(pw, lane, 32 * c);
#pragma unroll
        for (int t = 0; t < 8; ++t) {
            v16h vfrag = load_frag16(cV + 16 * t * KS, lane, 32 * c);
            O[t] = __builtin_amdgcn_wmma_f32_16x16x32_f16(
                       false, pa, false, vfrag, (short)0, O[t], false, false);
        }
    }
}

__global__ __launch_bounds__(256, 1)
void ExactAttention_flash_wmma_kernel(const float* __restrict__ q,
                                      const float* __restrict__ k,
                                      const float* __restrict__ v,
                                      float* __restrict__ out) {
    extern __shared__ __align__(16) char smem_raw[];
    _Float16* base = (_Float16*)smem_raw;
    // Compile-time-constant LDS partitions (keeps ds_* address-space inference)
    _Float16* sK0 = base;
    _Float16* sK1 = base + TILE_H;
    _Float16* sV0 = base + 2 * TILE_H;
    _Float16* sV1 = base + 3 * TILE_H;
    _Float16* sP  = base + 4 * TILE_H;     // 8 waves x [16 rows][KS]

    const int tid  = threadIdx.x;
    const int lane = tid & 31;
    const int wv   = tid >> 5;

    const int bid = blockIdx.x;
    const int nt  = bid & 15;          // row tile within sequence
    const int hh  = (bid >> 4) & 15;   // head
    const int bb  = bid >> 8;          // batch
    const int n0  = nt * BN;

    const int nl = lane & 15;
    const int hl = (lane >> 4) & 1;

    // ---- Per-thread staging identities ----
    const int skey  = tid >> 1;
    const int sdh   = (tid & 1) * 64;
    const int kpair = tid & 63;
    const int g     = tid >> 6;
    const float* kg0 = k + ((size_t)(bb * N_ + skey) * H_ + hh) * D_ + sdh;
    const float* vg0 = v + ((size_t)(bb * N_ + 2 * kpair) * H_ + hh) * D_ + 32 * g;

    // ---- Load Q strip directly into A fragments ----
    const int qrow = n0 + 16 * wv + nl;
    const float* qptr = q + ((size_t)(bb * N_ + qrow) * H_ + hh) * D_;
    v16h Qa[4];
#pragma unroll
    for (int c = 0; c < 4; ++c) {
        const float* p = qptr + 32 * c + 8 * hl;
        float4 a0 = *(const float4*)(p);
        float4 a1 = *(const float4*)(p + 4);
        float4 b0 = *(const float4*)(p + 16);
        float4 b1 = *(const float4*)(p + 20);
        v16h f;
        f[0] = (_Float16)a0.x; f[1] = (_Float16)a0.y; f[2]  = (_Float16)a0.z; f[3]  = (_Float16)a0.w;
        f[4] = (_Float16)a1.x; f[5] = (_Float16)a1.y; f[6]  = (_Float16)a1.z; f[7]  = (_Float16)a1.w;
        f[8] = (_Float16)b0.x; f[9] = (_Float16)b0.y; f[10] = (_Float16)b0.z; f[11] = (_Float16)b0.w;
        f[12]= (_Float16)b1.x; f[13]= (_Float16)b1.y; f[14] = (_Float16)b1.z; f[15] = (_Float16)b1.w;
        Qa[c] = f;
    }

    // ---- Accumulators & online-softmax state ----
    const v8f vzero = {0.f,0.f,0.f,0.f,0.f,0.f,0.f,0.f};
    v8f O[8];
#pragma unroll
    for (int t = 0; t < 8; ++t) O[t] = vzero;
    float mrow[8], lrow[8];
#pragma unroll
    for (int r = 0; r < 8; ++r) { mrow[r] = -3.0e38f; lrow[r] = 0.f; }

    _Float16* pw = sP + wv * (16 * KS);

    // ---- Prologue: stage tile 0 into buffer 0 ----
    float4 kreg[16], va[8], vb[8];
    load_tile(kg0, vg0, kreg, va, vb);
    store_tile(sK0 + skey * KS + sdh, sV0, g, kpair, kreg, va, vb);
    __syncthreads();

    // ---- Main loop: 2 tiles per iteration so LDS buffers are static ----
    for (int kt = 0; kt < NT; kt += 2) {
        // ===== even tile kt: compute from buf0, stage kt+1 -> buf1 =====
        load_tile(kg0 + (size_t)(kt + 1) * STEP,
                  vg0 + (size_t)(kt + 1) * STEP, kreg, va, vb);
        if (kt + 2 < NT) {
            const float* pk = kg0 + (size_t)(kt + 2) * STEP;
            __builtin_prefetch(pk, 0, 0);
            __builtin_prefetch(pk + 32, 0, 0);
            const float* pv = vg0 + (size_t)(kt + 2) * STEP;
            __builtin_prefetch(pv, 0, 0);
            __builtin_prefetch(pv + H_ * D_, 0, 0);
        }
        process_tile(sK0, sV0, pw, lane, nl, hl, Qa, O, mrow, lrow);
        store_tile(sK1 + skey * KS + sdh, sV1, g, kpair, kreg, va, vb);
        __syncthreads();

        // ===== odd tile kt+1: compute from buf1, stage kt+2 -> buf0 =====
        if (kt + 2 < NT) {
            load_tile(kg0 + (size_t)(kt + 2) * STEP,
                      vg0 + (size_t)(kt + 2) * STEP, kreg, va, vb);
            const float* pk = kg0 + (size_t)(kt + 3) * STEP;
            __builtin_prefetch(pk, 0, 0);
            __builtin_prefetch(pk + 32, 0, 0);
            const float* pv = vg0 + (size_t)(kt + 3) * STEP;
            __builtin_prefetch(pv, 0, 0);
            __builtin_prefetch(pv + H_ * D_, 0, 0);
        }
        process_tile(sK1, sV1, pw, lane, nl, hl, Qa, O, mrow, lrow);
        if (kt + 2 < NT)
            store_tile(sK0 + skey * KS + sdh, sV0, g, kpair, kreg, va, vb);
        __syncthreads();
    }

    // ---- Epilogue: divide by denominator, store [b,h,n,d] fp32 ----
    float* op = out + ((size_t)(bb * H_ + hh) * N_ + n0 + 16 * wv) * D_;
#pragma unroll
    for (int r = 0; r < 8; ++r) {
        const float inv = 1.0f / lrow[r];
        const int row = r + 8 * hl;
#pragma unroll
        for (int t = 0; t < 8; ++t)
            op[(size_t)row * D_ + 16 * t + nl] = O[t][r] * inv;
    }
}

extern "C" void kernel_launch(void* const* d_in, const int* in_sizes, int n_in,
                              void* d_out, int out_size, void* d_ws, size_t ws_size,
                              hipStream_t stream) {
    const float* q = (const float*)d_in[0];
    const float* k = (const float*)d_in[1];
    const float* v = (const float*)d_in[2];
    float* out = (float*)d_out;

    (void)in_sizes; (void)n_in; (void)out_size; (void)d_ws; (void)ws_size;

    hipFuncSetAttribute((const void*)ExactAttention_flash_wmma_kernel,
                        hipFuncAttributeMaxDynamicSharedMemorySize, SMEM_BYTES);

    dim3 grid(B_ * H_ * (N_ / BN));   // 512
    dim3 block(256);                  // 8 wave32s
    ExactAttention_flash_wmma_kernel<<<grid, block, SMEM_BYTES, stream>>>(q, k, v, out);
}
